// Transformer_71768903516412
// MI455X (gfx1250) — compile-verified, matrix operand order checked by reading, not run on hardware
//
#include <hip/hip_runtime.h>
#include <hip/hip_bf16.h>
#include <math.h>

// ---------------------------------------------------------------------------
// Problem constants (reference: N=4096 tokens, D=128, H=8 heads, HD=16, FF=512, L=2)
// ---------------------------------------------------------------------------
#define TN   4096
#define TD   128
#define TH_  8
#define THD  16
#define TFF  512
#define TL   2
#define NORMALIZER 0.25f      // 1/sqrt(16)
#define LN_EPS 1e-3f
#define L2_EPS 1e-12f

typedef __attribute__((ext_vector_type(16))) _Float16 v16h;
typedef __attribute__((ext_vector_type(8)))  float    v8f;

#define DEV __device__ __forceinline__

// ---------------------------------------------------------------------------
// WMMA helpers (gfx1250, wave32). 16x16x32 f16 -> f32.
// A-fragment: lane L holds row (L&15); elements 0..7 -> K=kb..kb+7,
// elements 8..15 -> K=kb+16..kb+23, kb=(L>>4)*8.  (ISA 7.12.2)
// B-fragment mirrors A with columns striped (lane holds col L&15), so a
// transposed (N-major) source loads contiguously, exactly like A.
// C/D fragment: lane L, VGPR r -> M = 8*(L>>4)+r, N = (L&15).
// ---------------------------------------------------------------------------
DEV v8f wmma16(v16h a, v16h b, v8f c) {
  return __builtin_amdgcn_wmma_f32_16x16x32_f16(
      false, a, false, b, (short)0, c, false, false);
}

// A[m][k] from row-major tile base (ld = row stride), zero-pad k>=kvalid
DEV v16h load_fragA(const _Float16* base, int ld, int kvalid) {
  int lane = threadIdx.x & 31;
  int r = lane & 15, kb = (lane >> 4) * 8;
  const _Float16* row = base + r * ld;
  v16h f;
#pragma unroll
  for (int i = 0; i < 8; ++i) {
    int k0 = kb + i, k1 = kb + 16 + i;
    f[i]     = (k0 < kvalid) ? row[k0] : (_Float16)0.f;
    f[8 + i] = (k1 < kvalid) ? row[k1] : (_Float16)0.f;
  }
  return f;
}

// B[k][n] = src[n][k] (N-major / transposed source), ld = row stride of src
DEV v16h load_fragBT(const _Float16* base, int ld, int kvalid) {
  int lane = threadIdx.x & 31;
  int n = lane & 15, kb = (lane >> 4) * 8;
  const _Float16* row = base + n * ld;
  v16h f;
#pragma unroll
  for (int i = 0; i < 8; ++i) {
    int k0 = kb + i, k1 = kb + 16 + i;
    f[i]     = (k0 < kvalid) ? row[k0] : (_Float16)0.f;
    f[8 + i] = (k1 < kvalid) ? row[k1] : (_Float16)0.f;
  }
  return f;
}

DEV float softplusf(float x) { return x > 20.f ? x : log1pf(__expf(x)); }

// CDNA5 async global->LDS copy (ASYNCcnt path), 16 bytes per lane.
DEV void async_g2l_b128(uint32_t lds_byte_off, uint64_t gaddr) {
  asm volatile("global_load_async_to_lds_b128 %0, %1, off"
               :: "v"(lds_byte_off), "v"(gaddr) : "memory");
}
template <int NCNT> DEV void wait_async() {
  asm volatile("s_wait_asynccnt %0" :: "i"(NCNT) : "memory");
}

// ---------------------------------------------------------------------------
// Block-tiled WMMA GEMM: C[M,Nn] = A_h[M,K] @ Bt_h[Nn,K]^T (+bias) (+softplus)
// Block = 256 threads = 8 waves -> 128x64 output tile.
// Double-buffered: the 64x32 f16 B panel for chunk k+1 is async-DMA'd into
// LDS while chunk k's WMMAs run; s_wait_asynccnt 1 (in-order completion)
// guarantees the panel being consumed has landed. Each wave reuses one A
// fragment across 4 WMMAs.
// ---------------------------------------------------------------------------
#define GBM 128
#define GBN 64
#define LDB 40   // padded LDS row (32 halfs + 8 pad) to spread banks

__global__ void xfm_gemm16(const _Float16* __restrict__ A, const _Float16* __restrict__ Bt,
                           const float* __restrict__ bias, float* __restrict__ C,
                           int M, int Nn, int K, int act) {
  int wave = threadIdx.x >> 5;
  int lane = threadIdx.x & 31;
  int row0 = blockIdx.x * GBM + wave * 16;
  int n0   = blockIdx.y * GBN;
  __shared__ _Float16 sB[2][GBN * LDB];

  // staging assignment: thread t moves 16B: B panel row sn, 8-half chunk sc
  int sn = threadIdx.x >> 2;
  int sc = threadIdx.x & 3;
  uint32_t ldsoff0 = (uint32_t)(uintptr_t)&sB[0][sn * LDB + sc * 8];
  uint32_t ldsoff1 = (uint32_t)(uintptr_t)&sB[1][sn * LDB + sc * 8];
  const _Float16* bsrc = Bt + (size_t)(n0 + sn) * K + sc * 8;

  int nk = K >> 5;
  // prologue: chunk 0 -> buffer 0
  async_g2l_b128(ldsoff0, (uint64_t)(uintptr_t)bsrc);

  v8f acc[4] = {};
  for (int ch = 0; ch < nk; ++ch) {
    int cur = ch & 1;
    if (ch + 1 < nk) {  // pipeline: DMA next panel into the other buffer
      async_g2l_b128(cur ? ldsoff0 : ldsoff1,
                     (uint64_t)(uintptr_t)(bsrc + (ch + 1) * 32));
      __builtin_prefetch(A + (size_t)row0 * K + (ch + 1) * 32, 0, 3);
    }
    v16h a = load_fragA(A + (size_t)row0 * K + ch * 32, K, 32);
    if (ch + 1 < nk) wait_async<1>(); else wait_async<0>();
    __syncthreads();   // all waves' copies of buffer `cur` have landed
    const _Float16* sbase = &sB[cur][0];
#pragma unroll
    for (int c = 0; c < 4; ++c) {
      v16h b = load_fragBT(sbase + (c * 16) * LDB, LDB, 32);
      acc[c] = wmma16(a, b, acc[c]);
    }
    __syncthreads();   // done reading `cur` before it is overwritten next round
  }
  int rowb = row0 + ((lane >> 4) << 3);
#pragma unroll
  for (int c = 0; c < 4; ++c) {
    int col = n0 + c * 16 + (lane & 15);
    float bv = bias ? bias[col] : 0.f;
#pragma unroll
    for (int r = 0; r < 8; ++r) {
      float v = acc[c][r] + bv;
      if (act == 1) v = softplusf(v);
      C[(size_t)(rowb + r) * Nn + col] = v;
    }
  }
}

// ---------------------------------------------------------------------------
// counter[i] = sum_j relu(cn_i . cn_j): block (8 waves) per 16-row tile,
// waves stride over key tiles; WMMA scores, relu+row-sum on f32 accumulator,
// shfl butterfly across the 16-lane C-fragment group, LDS combine.
// ---------------------------------------------------------------------------
__global__ void xfm_counter_rowsum(const _Float16* __restrict__ cnh, float* __restrict__ craw) {
  int it = blockIdx.x;
  int wave = threadIdx.x >> 5;
  int lane = threadIdx.x & 31;
  __shared__ float srs[16];
  if (threadIdx.x < 16) srs[threadIdx.x] = 0.f;
  __syncthreads();

  const _Float16* arow = cnh + (size_t)it * 16 * TD;
  v16h afr[4];
#pragma unroll
  for (int kk = 0; kk < 4; ++kk) afr[kk] = load_fragA(arow + kk * 32, TD, 32);

  float rs[8];
#pragma unroll
  for (int r = 0; r < 8; ++r) rs[r] = 0.f;

  for (int jt = wave; jt < TN / 16; jt += 8) {
    const _Float16* bbase = cnh + (size_t)jt * 16 * TD;  // B = cn^T (rows contiguous)
    v8f c = {};
#pragma unroll
    for (int kk = 0; kk < 4; ++kk) {
      v16h b = load_fragBT(bbase + kk * 32, TD, 32);
      c = wmma16(afr[kk], b, c);
    }
#pragma unroll
    for (int r = 0; r < 8; ++r) rs[r] += fmaxf(c[r], 0.f);
  }
#pragma unroll
  for (int m = 1; m < 16; m <<= 1)
#pragma unroll
    for (int r = 0; r < 8; ++r) rs[r] += __shfl_xor(rs[r], m, 32);
  if ((lane & 15) == 0) {
    int rb = (lane >> 4) * 8;
#pragma unroll
    for (int r = 0; r < 8; ++r) atomicAdd(&srs[rb + r], rs[r]);
  }
  __syncthreads();
  if (threadIdx.x < 16) craw[it * 16 + threadIdx.x] = srs[threadIdx.x];
}

// ---------------------------------------------------------------------------
// Flash attention: block = 8 waves, each wave owns one (16-query tile, head),
// 32 keys per iteration: 2 score WMMAs (K=HD=16, padded) + 1 full-K PV WMMA.
// P staged via a per-wave LDS slice to reshape C-layout -> A-layout; V read
// contiguously from a transposed copy. Uniform trip count keeps barriers legal.
// ---------------------------------------------------------------------------
__global__ void xfm_attention(const _Float16* __restrict__ qh, const _Float16* __restrict__ kvh,
                              const _Float16* __restrict__ vvt, _Float16* __restrict__ headsh) {
  int wave = threadIdx.x >> 5;
  int lane = threadIdx.x & 31;
  int qt = blockIdx.x * 8 + wave;
  int h  = blockIdx.y;
  __shared__ _Float16 spAll[8][16 * 32];
  _Float16* sp = spAll[wave];

  v16h aq = load_fragA(qh + (size_t)qt * 16 * TD + h * THD, TD, THD);
  v8f o = {};
  float mrun[8], lrun[8];
#pragma unroll
  for (int r = 0; r < 8; ++r) { mrun[r] = -1e30f; lrun[r] = 0.f; }

  int rb = (lane >> 4) * 8, col = lane & 15;
  for (int jt = 0; jt < TN / 32; ++jt) {
    const _Float16* kb0 = kvh + (size_t)(jt * 32) * (2 * TD) + h * THD;  // key rows
    v16h bk0 = load_fragBT(kb0, 2 * TD, THD);
    v16h bk1 = load_fragBT(kb0 + 16 * (2 * TD), 2 * TD, THD);
    v8f s0 = {}, s1 = {};
    s0 = wmma16(aq, bk0, s0);
    s1 = wmma16(aq, bk1, s1);

#pragma unroll
    for (int r = 0; r < 8; ++r) {
      float a0 = s0[r] * NORMALIZER;
      float a1 = s1[r] * NORMALIZER;
      float mx = fmaxf(a0, a1);
#pragma unroll
      for (int m = 1; m < 16; m <<= 1) mx = fmaxf(mx, __shfl_xor(mx, m, 32));
      float mnew = fmaxf(mrun[r], mx);
      float alpha = __expf(mrun[r] - mnew);
      float p0 = __expf(a0 - mnew);
      float p1 = __expf(a1 - mnew);
      float ps = p0 + p1;
#pragma unroll
      for (int m = 1; m < 16; m <<= 1) ps += __shfl_xor(ps, m, 32);
      lrun[r] = lrun[r] * alpha + ps;
      mrun[r] = mnew;
      o[r] *= alpha;
      sp[(rb + r) * 32 + col]      = (_Float16)p0;
      sp[(rb + r) * 32 + 16 + col] = (_Float16)p1;
    }
    __syncthreads();
    v16h ap = load_fragA(sp, 32, 32);
    v16h bv = load_fragBT(vvt + (size_t)(h * THD) * TN + jt * 32, TN, 32);
    o = wmma16(ap, bv, o);
    __syncthreads();
  }
#pragma unroll
  for (int r = 0; r < 8; ++r) {
    float val = o[r] / lrun[r];
    headsh[(size_t)(qt * 16 + rb + r) * TD + h * THD + col] = (_Float16)val;
  }
}

// ---------------------------------------------------------------------------
// Elementwise / normalization kernels (f32 math, f16 side outputs for WMMA)
// ---------------------------------------------------------------------------
__global__ void xfm_cvt_f16(const float* __restrict__ in, _Float16* __restrict__ out, int n) {
  int i = blockIdx.x * blockDim.x + threadIdx.x;
  if (i < n) out[i] = (_Float16)in[i];
}

// f32 [K][Nn] row-major -> f16 transposed [Nn][K]
__global__ void xfm_cvt_f16_t(const float* __restrict__ in, _Float16* __restrict__ out,
                              int K, int Nn) {
  int i = blockIdx.x * blockDim.x + threadIdx.x;
  if (i >= K * Nn) return;
  int k = i / Nn, n = i - k * Nn;
  out[(size_t)n * K + k] = (_Float16)in[i];
}

// per-row L2 normalize (wave per row)
__global__ void xfm_l2norm(const float* __restrict__ x, _Float16* __restrict__ outh) {
  int row = blockIdx.x * (blockDim.x >> 5) + (threadIdx.x >> 5);
  int lane = threadIdx.x & 31;
  const float* xr = x + (size_t)row * TD;
  float v[4], s = 0.f;
#pragma unroll
  for (int i = 0; i < 4; ++i) { v[i] = xr[lane + 32 * i]; s += v[i] * v[i]; }
#pragma unroll
  for (int m = 16; m >= 1; m >>= 1) s += __shfl_xor(s, m, 32);
  float inv = rsqrtf(fmaxf(s, L2_EPS));
#pragma unroll
  for (int i = 0; i < 4; ++i) outh[(size_t)row * TD + lane + 32 * i] = (_Float16)(v[i] * inv);
}

// y = LN(a + res)*g + b  (wave per row); outf/outh nullable
__global__ void xfm_layernorm(const float* __restrict__ a, const float* __restrict__ res,
                              const float* __restrict__ g, const float* __restrict__ bb,
                              float* __restrict__ outf, _Float16* __restrict__ outh) {
  int row = blockIdx.x * (blockDim.x >> 5) + (threadIdx.x >> 5);
  int lane = threadIdx.x & 31;
  float v[4], s = 0.f;
#pragma unroll
  for (int i = 0; i < 4; ++i) {
    int c = lane + 32 * i;
    float t = a[(size_t)row * TD + c] + (res ? res[(size_t)row * TD + c] : 0.f);
    v[i] = t; s += t;
  }
#pragma unroll
  for (int m = 16; m >= 1; m >>= 1) s += __shfl_xor(s, m, 32);
  float mean = s * (1.f / TD);
  float var = 0.f;
#pragma unroll
  for (int i = 0; i < 4; ++i) { float d = v[i] - mean; var += d * d; }
#pragma unroll
  for (int m = 16; m >= 1; m >>= 1) var += __shfl_xor(var, m, 32);
  float inv = rsqrtf(var * (1.f / TD) + LN_EPS);
#pragma unroll
  for (int i = 0; i < 4; ++i) {
    int c = lane + 32 * i;
    float y = (v[i] - mean) * inv * g[c] + bb[c];
    if (outf) outf[(size_t)row * TD + c] = y;
    if (outh) outh[(size_t)row * TD + c] = (_Float16)y;
  }
}

// per-column standardization over token axis (block per column).
// vvt != nullptr: also write transposed f16 copy of columns >= TD (values half).
__global__ void xfm_colnorm(const float* __restrict__ in, _Float16* __restrict__ outh,
                            _Float16* __restrict__ vvt, int C) {
  int col = blockIdx.x;
  int t = threadIdx.x;
  float s = 0.f, s2 = 0.f;
  for (int r = t; r < TN; r += 256) {
    float v = in[(size_t)r * C + col];
    s += v; s2 += v * v;
  }
  __shared__ float sh[256], sh2[256];
  sh[t] = s; sh2[t] = s2; __syncthreads();
  for (int o = 128; o > 0; o >>= 1) {
    if (t < o) { sh[t] += sh[t + o]; sh2[t] += sh2[t + o]; }
    __syncthreads();
  }
  float mean = sh[0] * (1.f / TN);
  float var = sh2[0] * (1.f / TN) - mean * mean;
  float inv = rsqrtf(fmaxf(var, 1e-30f));
  for (int r = t; r < TN; r += 256) {
    float y = (in[(size_t)r * C + col] - mean) * inv;
    outh[(size_t)r * C + col] = (_Float16)y;
    if (vvt && col >= TD) vvt[(size_t)(col - TD) * TN + r] = (_Float16)y;
  }
}

// cat_h[i, 0:128] = f16(x[i]);  cat_h[i, 128:256] = f16(softplus(craw[i]*ce_w + ce_b))
__global__ void xfm_build_cat(const float* __restrict__ xf, const float* __restrict__ craw,
                              const float* __restrict__ cew, const float* __restrict__ ceb,
                              _Float16* __restrict__ cath) {
  int i = blockIdx.x * blockDim.x + threadIdx.x;
  if (i >= TN * TD) return;
  int row = i >> 7, d = i & 127;
  cath[(size_t)row * 256 + d] = (_Float16)xf[i];
  float cv = softplusf(craw[row] * cew[d] + ceb[d]);
  cath[(size_t)row * 256 + 128 + d] = (_Float16)cv;
}

// ---------------------------------------------------------------------------
// Host orchestration
// ---------------------------------------------------------------------------
extern "C" void kernel_launch(void* const* d_in, const int* in_sizes, int n_in,
                              void* d_out, int out_size, void* d_ws, size_t ws_size,
                              hipStream_t stream) {
  const float* x0   = (const float*)d_in[0];
  const float* Wq   = (const float*)d_in[1];
  const float* bq   = (const float*)d_in[2];
  const float* Wkv  = (const float*)d_in[3];
  const float* bkv  = (const float*)d_in[4];
  const float* Wo   = (const float*)d_in[5];
  const float* bo   = (const float*)d_in[6];
  const float* ce_w = (const float*)d_in[7];
  const float* ce_b = (const float*)d_in[8];
  const float* cm_w = (const float*)d_in[9];
  const float* cn_g = (const float*)d_in[10];
  const float* cn_b = (const float*)d_in[11];
  const float* an_g = (const float*)d_in[12];
  const float* an_b = (const float*)d_in[13];
  const float* W1   = (const float*)d_in[14];
  const float* b1   = (const float*)d_in[15];
  const float* W2   = (const float*)d_in[16];
  const float* b2   = (const float*)d_in[17];
  const float* fn_g = (const float*)d_in[18];
  const float* fn_b = (const float*)d_in[19];

  // ---- workspace bump allocator ----
  char* p = (char*)d_ws;
  auto alloc = [&](size_t bytes) -> void* {
    void* r = (void*)p;
    p += (bytes + 255) & ~(size_t)255;
    return r;
  };
  float*    xf     = (float*)alloc((size_t)TN * TD * 4);
  _Float16* xh     = (_Float16*)alloc((size_t)TN * TD * 2);
  _Float16* cnh    = (_Float16*)alloc((size_t)TN * TD * 2);
  float*    craw   = (float*)alloc((size_t)TN * 4);
  _Float16* vh     = (_Float16*)alloc((size_t)TN * TD * 2);
  _Float16* kvh    = (_Float16*)alloc((size_t)TN * 2 * TD * 2);
  _Float16* vvt    = (_Float16*)alloc((size_t)TD * TN * 2);   // values transposed [128][N]
  _Float16* qh     = (_Float16*)alloc((size_t)TN * TD * 2);
  _Float16* headsh = (_Float16*)alloc((size_t)TN * TD * 2);
  float*    tf     = (float*)alloc((size_t)TN * TFF * 4);     // reused f32 temp
  _Float16* thb    = (_Float16*)alloc((size_t)TN * TFF * 2);  // cat (N,256) / f1h (N,512)
  // transposed f16 weights [Nout][K]
  _Float16* wqt  = (_Float16*)alloc((size_t)TD * TD * 2);
  _Float16* wkvt = (_Float16*)alloc((size_t)2 * TD * TD * 2);
  _Float16* wot  = (_Float16*)alloc((size_t)TD * TD * 2);
  _Float16* cmt  = (_Float16*)alloc((size_t)TD * 2 * TD * 2);
  _Float16* w1t  = (_Float16*)alloc((size_t)TFF * TD * 2);
  _Float16* w2t  = (_Float16*)alloc((size_t)TD * TFF * 2);

  auto cvt = [&](const float* src, _Float16* dst, int n) {
    xfm_cvt_f16<<<(n + 255) / 256, 256, 0, stream>>>(src, dst, n);
  };
  auto cvt_t = [&](const float* src, _Float16* dst, int K, int Nn) {
    xfm_cvt_f16_t<<<(K * Nn + 255) / 256, 256, 0, stream>>>(src, dst, K, Nn);
  };
  auto gemm = [&](const _Float16* A, const _Float16* Bt, const float* bias, float* C,
                  int M, int Nn, int K, int act) {
    dim3 g(M / GBM, Nn / GBN);
    xfm_gemm16<<<g, 256, 0, stream>>>(A, Bt, bias, C, M, Nn, K, act);
  };

  // x = target_sequence
  hipMemcpyAsync(xf, x0, (size_t)TN * TD * 4, hipMemcpyDeviceToDevice, stream);
  cvt(xf, xh, TN * TD);

  for (int l = 0; l < TL; ++l) {
    const float* lWq  = Wq + (size_t)l * TD * TD;
    const float* lbq  = bq + (size_t)l * TD;
    const float* lWkv = Wkv + (size_t)l * TD * 2 * TD;
    const float* lbkv = bkv + (size_t)l * 2 * TD;
    const float* lWo  = Wo + (size_t)l * TD * TD;
    const float* lbo  = bo + (size_t)l * TD;
    const float* lcew = ce_w + (size_t)l * TD;
    const float* lceb = ce_b + (size_t)l * TD;
    const float* lcm  = cm_w + (size_t)l * 2 * TD * TD;
    const float* lcng = cn_g + (size_t)l * TD, *lcnb = cn_b + (size_t)l * TD;
    const float* lang = an_g + (size_t)l * TD, *lanb = an_b + (size_t)l * TD;
    const float* lW1  = W1 + (size_t)l * TD * TFF;
    const float* lb1  = b1 + (size_t)l * TFF;
    const float* lW2  = W2 + (size_t)l * TFF * TD;
    const float* lb2  = b2 + (size_t)l * TD;
    const float* lfng = fn_g + (size_t)l * TD, *lfnb = fn_b + (size_t)l * TD;

    // transposed f16 weight copies for WMMA B operands
    cvt_t(lWq,  wqt,  TD, TD);
    cvt_t(lWkv, wkvt, TD, 2 * TD);
    cvt_t(lWo,  wot,  TD, TD);
    cvt_t(lcm,  cmt,  2 * TD, TD);
    cvt_t(lW1,  w1t,  TD, TFF);
    cvt_t(lW2,  w2t,  TFF, TD);

    // ---- counter path ----
    xfm_l2norm<<<TN / 8, 256, 0, stream>>>(xf, cnh);
    xfm_counter_rowsum<<<TN / 16, 256, 0, stream>>>(cnh, craw);
    xfm_build_cat<<<(TN * TD + 255) / 256, 256, 0, stream>>>(xf, craw, lcew, lceb, thb);
    gemm(thb, cmt, nullptr, tf, TN, TD, 2 * TD, 0);                    // [x|counter] @ cm_w
    xfm_layernorm<<<TN / 8, 256, 0, stream>>>(tf, nullptr, lcng, lcnb, nullptr, vh);

    // ---- attention ----
    gemm(vh, wkvt, lbkv, tf, TN, 2 * TD, TD, 0);                       // v @ Wkv + bkv
    xfm_colnorm<<<2 * TD, 256, 0, stream>>>(tf, kvh, vvt, 2 * TD);
    gemm(xh, wqt, lbq, tf, TN, TD, TD, 0);                             // x @ Wq + bq
    xfm_colnorm<<<TD, 256, 0, stream>>>(tf, qh, nullptr, TD);
    dim3 ag(TN / 16 / 8, TH_);
    xfm_attention<<<ag, 256, 0, stream>>>(qh, kvh, vvt, headsh);
    gemm(headsh, wot, lbo, tf, TN, TD, TD, 0);                         // heads @ Wo + bo
    xfm_layernorm<<<TN / 8, 256, 0, stream>>>(tf, xf, lang, lanb, xf, xh);  // x = LN(x+att)

    // ---- FFN ----
    gemm(xh, w1t, lb1, tf, TN, TFF, TD, 1);                            // softplus(x@W1+b1)
    cvt(tf, thb, TN * TFF);
    gemm(thb, w2t, lb2, tf, TN, TD, TFF, 0);                           // @W2 + b2
    xfm_layernorm<<<TN / 8, 256, 0, stream>>>(tf, xf, lfng, lfnb, xf, xh);  // x = LN(x+ffn)
  }

  hipMemcpyAsync(d_out, xf, (size_t)TN * TD * 4, hipMemcpyDeviceToDevice, stream);
}
